// MemoryOptimizedMLA_23055384445377
// MI455X (gfx1250) — matrix operand order repaired and checked by reading, not compiled, not run on hardware
//
#include <hip/hip_runtime.h>
#include <cstdint>
#include <cstddef>

#define D_MODEL 5120
#define N_HEADS 8
#define D_HEAD  640
#define D_ROPE  16
#define SPLIT   624
#define D_KV    128
#define BATCH   2
#define SEQ     2048
#define NROWS   (BATCH*SEQ)     // 4096

typedef __attribute__((ext_vector_type(16))) __bf16 v16bf;
typedef __attribute__((ext_vector_type(8)))  float  v8f;
typedef __attribute__((ext_vector_type(4)))  unsigned int v4u;
typedef __attribute__((ext_vector_type(8)))  int v8i;
typedef __attribute__((ext_vector_type(4)))  int v4i;

union FragU {
  v16bf v;
  uint4 q[2];
};

__device__ __forceinline__ unsigned short f2bf(float f) {
  return __builtin_bit_cast(unsigned short, (__bf16)f);   // native cvt, RNE
}
__device__ __forceinline__ unsigned int pk2(float a, float b) {
  return (unsigned int)f2bf(a) | ((unsigned int)f2bf(b) << 16);
}

// A fragment (16x32 bf16). Lane = row (mod 16); lane-half selects K+8;
// elements 0-7 -> K = kh+0..7, elements 8-15 -> K = 16+kh+0..7.
__device__ __forceinline__ v16bf load_afrag(const unsigned short* base, int stride, int lane) {
  const int r  = lane & 15;
  const int kh = (lane >> 4) << 3;
  const unsigned short* p = base + (size_t)r * stride + kh;
  FragU f;
  f.q[0] = *(const uint4*)(p);
  f.q[1] = *(const uint4*)(p + 16);
  return f.v;
}

// A fragment built from f32 source with on-the-fly bf16 conversion.
__device__ __forceinline__ v16bf load_afrag_f32(const float* base, int stride, int lane) {
  const int r  = lane & 15;
  const int kh = (lane >> 4) << 3;
  const float* p = base + (size_t)r * stride + kh;
  float4 x0 = *(const float4*)(p);
  float4 x1 = *(const float4*)(p + 4);
  float4 x2 = *(const float4*)(p + 16);
  float4 x3 = *(const float4*)(p + 20);
  v16bf r16;
  r16[0]  = (__bf16)x0.x; r16[1]  = (__bf16)x0.y; r16[2]  = (__bf16)x0.z; r16[3]  = (__bf16)x0.w;
  r16[4]  = (__bf16)x1.x; r16[5]  = (__bf16)x1.y; r16[6]  = (__bf16)x1.z; r16[7]  = (__bf16)x1.w;
  r16[8]  = (__bf16)x2.x; r16[9]  = (__bf16)x2.y; r16[10] = (__bf16)x2.z; r16[11] = (__bf16)x2.w;
  r16[12] = (__bf16)x3.x; r16[13] = (__bf16)x3.y; r16[14] = (__bf16)x3.z; r16[15] = (__bf16)x3.w;
  return r16;
}

// B fragment (32x16 bf16). Lane = column (mod 16); lane-half selects K+8;
// elements 0-7 -> K = ko+0..7, elements 8-15 -> K = 16+ko+0..7;
// source must be K-contiguous per column with `stride` between cols.
__device__ __forceinline__ v16bf load_bfrag(const unsigned short* base, int stride, int lane) {
  const int n  = lane & 15;
  const int ko = (lane >> 4) << 3;
  const unsigned short* p = base + (size_t)n * stride + ko;
  FragU f;
  f.q[0] = *(const uint4*)(p);
  f.q[1] = *(const uint4*)(p + 16);
  return f.v;
}

__device__ __forceinline__ v8f wmma_bf16(v16bf a, v16bf b, v8f c) {
  return __builtin_amdgcn_wmma_f32_16x16x32_bf16(false, a, false, b, (short)0, c,
                                                 false, false);
}

__device__ __forceinline__ unsigned lds_addr_of(const void* p) {
  return (unsigned)(unsigned long long)(__attribute__((address_space(3))) const void*)p;
}

// Issue one 2D TDM tile load (bf16 elements): tile_d0 contiguous elements per
// row, tile_d1 rows, tensor row stride in elements. D# groups per ISA 8.3/8.4.
// This toolchain exposes the 6-arg builtin (g0, g1, g2, g3, g4, cpol).
__device__ __forceinline__ void tdm_load_2d(const void* gsrc, unsigned lds_byte_addr,
                                            unsigned tile_d0, unsigned tile_d1,
                                            unsigned stride0_elems) {
  const unsigned long long ga = (unsigned long long)gsrc;
  v4u g0;
  g0.x = 1u;                                                  // count=1, user D#
  g0.y = lds_byte_addr;                                       // lds_addr
  g0.z = (unsigned)ga;                                        // global_addr lo
  g0.w = (unsigned)((ga >> 32) & 0x01ffffffu) | (2u << 30);   // ga hi | type=2
  v8i g1;
  g1[0] = (int)(1u << 16);                                    // data_size=1 (2B)
  g1[1] = (int)((tile_d0 & 0xffffu) << 16);                   // tensor_dim0 lo16
  g1[2] = (int)(((tile_d0 >> 16) & 0xffffu) |                 // tensor_dim0 hi
                ((tile_d1 & 0xffffu) << 16));                 // tensor_dim1 lo16
  g1[3] = (int)(((tile_d1 >> 16) & 0xffffu) |                 // tensor_dim1 hi
                ((tile_d0 & 0xffffu) << 16));                 // tile_dim0
  g1[4] = (int)(tile_d1 & 0xffffu);                           // tile_dim1 (dim2=0)
  g1[5] = (int)stride0_elems;                                 // tensor_dim0_stride lo32
  g1[6] = 0; g1[7] = 0;                                       // stride hi / dim1_stride
  v4i z4 = {0, 0, 0, 0};
  v8i z8 = {0, 0, 0, 0, 0, 0, 0, 0};
  __builtin_amdgcn_tensor_load_to_lds(g0, g1, z4, z4, z8, 0);
}

// ---------------------------------------------------------------------------
// Kernel W: out_w (f32 [k][n]) -> bf16 transposed [n][k] (B-fragment layout).
// grid = (160,160), block = 256 (32x8 tile threads).
// ---------------------------------------------------------------------------
__global__ void __launch_bounds__(256)
k_wprep(const float* __restrict__ outw, unsigned short* __restrict__ outw_bfT)
{
  __shared__ float tile[32][33];
  const int k0 = blockIdx.x * 32;
  const int n0 = blockIdx.y * 32;
  const int tx = threadIdx.x & 31, ty = threadIdx.x >> 5;
  #pragma unroll
  for (int i = 0; i < 4; ++i)
    tile[ty + i * 8][tx] = outw[(size_t)(k0 + ty + i * 8) * D_MODEL + n0 + tx];
  __syncthreads();
  #pragma unroll
  for (int i = 0; i < 4; ++i)
    outw_bfT[(size_t)(n0 + ty + i * 8) * D_MODEL + k0 + tx] = f2bf(tile[tx][ty + i * 8]);
}

// ---------------------------------------------------------------------------
// Kernel A: c_kv = h @ W_dkv + b ; c_q = h @ W_dq + b
// grid = (256 m-tiles, 4 col-groups of 64), block = 128 (4 waves)
// ---------------------------------------------------------------------------
__global__ void __launch_bounds__(128)
k_down_proj(const float* __restrict__ h,
            const float* __restrict__ Wdkv, const float* __restrict__ bdkv,
            const float* __restrict__ Wdq,  const float* __restrict__ bdq,
            float* __restrict__ ckv_f32,
            unsigned short* __restrict__ ckv_bf,
            unsigned short* __restrict__ cq_bf)
{
  __shared__ __align__(16) unsigned short Bt[64][32];
  const int m0   = blockIdx.x * 16;
  const int grp  = blockIdx.y;
  const bool is_q = grp >= 2;
  const int n0b  = (grp & 1) * 64;
  const float* W    = is_q ? Wdq : Wdkv;
  const float* bias = is_q ? bdq : bdkv;
  const int tid = threadIdx.x;
  const int wave = tid >> 5, lane = tid & 31;

  v8f acc = {};
  for (int k0 = 0; k0 < D_MODEL; k0 += 32) {
    {   // stage 32x64 f32 weight chunk -> bf16, transposed (K-contiguous)
      const int kr = tid & 31, cg = tid >> 5;
      const float* wp = W + (size_t)(k0 + kr) * D_KV + n0b + cg * 16;
      float wv[16];
      *(float4*)&wv[0]  = *(const float4*)(wp);
      *(float4*)&wv[4]  = *(const float4*)(wp + 4);
      *(float4*)&wv[8]  = *(const float4*)(wp + 8);
      *(float4*)&wv[12] = *(const float4*)(wp + 12);
      #pragma unroll
      for (int j = 0; j < 16; ++j) Bt[cg * 16 + j][kr] = f2bf(wv[j]);
    }
    __syncthreads();
    v16bf a = load_afrag_f32(h + (size_t)m0 * D_MODEL + k0, D_MODEL, lane);
    v16bf b = load_bfrag(&Bt[wave * 16][0], 32, lane);
    acc = wmma_bf16(a, b, acc);
    __syncthreads();
  }
  const int n  = lane & 15;
  const int mh = (lane >> 4) << 3;
  const float bn = bias[n0b + wave * 16 + n];
  #pragma unroll
  for (int v = 0; v < 8; ++v) {
    const float val = acc[v] + bn;
    const size_t idx = (size_t)(m0 + v + mh) * D_KV + n0b + wave * 16 + n;
    if (!is_q) { ckv_f32[idx] = val; ckv_bf[idx] = f2bf(val); }
    else       { cq_bf[idx] = f2bf(val); }
  }
}

// ---------------------------------------------------------------------------
// Kernel B: up-projections + RoPE.
// Column space: 960 tiles of 16 = [Q 0..319 | K 320..639 | V 640..959];
// 40 tiles per head, tile 39 is the 16-wide RoPE slice.
// q,k stored [b,s,h,640] bf16; v stored transposed [b*h, d, s] bf16.
// grid = (256, 240), block = 128.
// ---------------------------------------------------------------------------
__global__ void __launch_bounds__(128)
k_up_proj_rope(const unsigned short* __restrict__ ckv_bf,
               const unsigned short* __restrict__ cq_bf,
               const float* __restrict__ Wuk, const float* __restrict__ buk,
               const float* __restrict__ Wuv, const float* __restrict__ buv,
               const float* __restrict__ Wuq, const float* __restrict__ buq,
               const float* __restrict__ Wqr, const float* __restrict__ bqr,
               const float* __restrict__ Wkr, const float* __restrict__ bkr,
               unsigned short* __restrict__ qbf,
               unsigned short* __restrict__ kbf,
               unsigned short* __restrict__ vT,
               float* __restrict__ krot_out)
{
  __shared__ __align__(16) unsigned short Bt[4][16][32];
  const int m0  = blockIdx.x * 16;
  const int tid = threadIdx.x;
  const int wave = tid >> 5, lane = tid & 31;
  const int t      = blockIdx.y * 4 + wave;   // 0..959
  const int region = t / 320;                 // 0=Q 1=K 2=V
  const int tr     = t % 320;
  const int head   = tr / 40, sub = tr % 40;

  const float *W, *bias; int ldn, sn0; bool rope = false;
  if (region == 0) {
    if (sub < 39) { W = Wuq; bias = buq; ldn = N_HEADS * SPLIT;  sn0 = head * SPLIT + sub * 16; }
    else          { W = Wqr; bias = bqr; ldn = N_HEADS * D_ROPE; sn0 = head * D_ROPE; rope = true; }
  } else if (region == 1) {
    if (sub < 39) { W = Wuk; bias = buk; ldn = N_HEADS * SPLIT;  sn0 = head * SPLIT + sub * 16; }
    else          { W = Wkr; bias = bkr; ldn = N_HEADS * D_ROPE; sn0 = head * D_ROPE; rope = true; }
  } else        { W = Wuv; bias = buv; ldn = N_HEADS * D_HEAD; sn0 = head * D_HEAD + sub * 16; }
  const unsigned short* cbf = (region == 0) ? cq_bf : ckv_bf;

  v8f acc = {};
  for (int k0 = 0; k0 < D_KV; k0 += 32) {
    {   // per-wave staging of its own 32x16 weight chunk (transposed, bf16)
      const float* wp = W + (size_t)(k0 + lane) * ldn + sn0;
      float wv[16];
      *(float4*)&wv[0]  = *(const float4*)(wp);
      *(float4*)&wv[4]  = *(const float4*)(wp + 4);
      *(float4*)&wv[8]  = *(const float4*)(wp + 8);
      *(float4*)&wv[12] = *(const float4*)(wp + 12);
      #pragma unroll
      for (int j = 0; j < 16; ++j) Bt[wave][j][lane] = f2bf(wv[j]);
    }
    __syncthreads();
    v16bf a = load_afrag(cbf + (size_t)m0 * D_KV + k0, D_KV, lane);
    v16bf b = load_bfrag(&Bt[wave][0][0], 32, lane);
    acc = wmma_bf16(a, b, acc);
    __syncthreads();
  }

  const int n  = lane & 15;
  const int mh = (lane >> 4) << 3;
  const float bn = bias[sn0 + n];
  float c[8];
  #pragma unroll
  for (int v = 0; v < 8; ++v) c[v] = acc[v] + bn;

  if (rope) {
    const int jm = n & 7;
    const float inv_freq = __powf(1000.0f, -(float)jm / 8.0f);
    #pragma unroll
    for (int v = 0; v < 8; ++v) {
      const int gm = m0 + v + mh;
      const int s  = gm & (SEQ - 1);
      const float ang = ((float)s / 40.0f) * inv_freq;
      const float cs = __cosf(ang), si = __sinf(ang);
      const float other = __shfl_xor(c[v], 8, 32);    // partner column n^8
      c[v] = c[v] * cs + ((n < 8) ? -other : other) * si;
    }
    if (region == 1) {        // k_rot (post-RoPE, f32) is a kernel output
      #pragma unroll
      for (int v = 0; v < 8; ++v)
        krot_out[(size_t)(m0 + v + mh) * (N_HEADS * D_ROPE) + head * D_ROPE + n] = c[v];
    }
  }

  if (region == 2) {          // V: store transposed [bh, d, s], 8 rows packed
    const int b  = m0 / SEQ;
    const int s0 = (m0 & (SEQ - 1)) + mh;
    const int d  = sub * 16 + n;
    uint4 u;
    u.x = pk2(c[0], c[1]); u.y = pk2(c[2], c[3]);
    u.z = pk2(c[4], c[5]); u.w = pk2(c[6], c[7]);
    *(uint4*)(vT + ((size_t)((b * N_HEADS + head) * D_HEAD + d)) * SEQ + s0) = u;
  } else {
    unsigned short* dst = (region == 0) ? qbf : kbf;
    const int d = (sub < 39) ? (sub * 16 + n) : (SPLIT + n);
    #pragma unroll
    for (int v = 0; v < 8; ++v)
      dst[(size_t)(m0 + v + mh) * D_MODEL + head * D_HEAD + d] = f2bf(c[v]);
  }
}

// ---------------------------------------------------------------------------
// Kernel C: flash attention with TDM-double-buffered K/V tiles in LDS.
// grid = (S/16, H, B), block = 128 (4 waves); each wave owns a 160-wide
// d-slice (5 A-frag chunks, 10 O tiles). K tile 32x640, V tile 640x32.
// ---------------------------------------------------------------------------
__global__ void __launch_bounds__(128)
k_attention(const unsigned short* __restrict__ qbf,
            const unsigned short* __restrict__ kbf,
            const unsigned short* __restrict__ vT,
            unsigned short* __restrict__ attn_bf)
{
  __shared__ __align__(16) unsigned short Kt[2][32][640];   // 80 KB
  __shared__ __align__(16) unsigned short Vt[2][640][32];   // 80 KB
  __shared__ __align__(16) float sred[4][2][8][32];         // 8 KB
  __shared__ __align__(16) unsigned short Pb[16][32];
  __shared__ float m_run[16], l_run[16], alpha_s[16];

  const int qt = blockIdx.x, head = blockIdx.y, b = blockIdx.z;
  const int tid = threadIdx.x;
  const int wave = tid >> 5, lane = tid & 31;
  const int q0 = qt * 16;
  const int d0 = wave * 160;

  if (tid < 16) { m_run[tid] = -1e30f; l_run[tid] = 0.0f; }

  const unsigned short* kband = kbf + (size_t)b * SEQ * D_MODEL + head * D_HEAD;
  const unsigned short* vband = vT + (size_t)((b * N_HEADS + head) * D_HEAD) * SEQ;

  if (wave == 0) {    // prologue: prefetch tile 0
    tdm_load_2d(kband, lds_addr_of(&Kt[0][0][0]), 640, 32, D_MODEL);
    tdm_load_2d(vband, lds_addr_of(&Vt[0][0][0]), 32, 640, SEQ);
  }

  const unsigned short* qrow = qbf + (size_t)(b * SEQ + q0) * D_MODEL + head * D_HEAD + d0;
  v16bf aq[5];
  #pragma unroll
  for (int c = 0; c < 5; ++c) aq[c] = load_afrag(qrow + c * 32, D_MODEL, lane);

  v8f o[10] = {};

  for (int kt = 0; kt < SEQ / 32; ++kt) {
    const int buf = kt & 1;
    __syncthreads();                       // everyone done reading buf^1
    if (wave == 0) {
      if (kt + 1 < SEQ / 32) {             // prefetch next tile into buf^1
        const size_t knext = (size_t)(kt + 1) * 32;
        tdm_load_2d(kband + knext * D_MODEL, lds_addr_of(&Kt[buf ^ 1][0][0]),
                    640, 32, D_MODEL);
        tdm_load_2d(vband + knext, lds_addr_of(&Vt[buf ^ 1][0][0]),
                    32, 640, SEQ);
        __builtin_amdgcn_s_wait_tensorcnt(2);   // current buf's 2 loads done
      } else {
        __builtin_amdgcn_s_wait_tensorcnt(0);
      }
    }
    __syncthreads();                       // buf visible to all waves

    // --- partial scores over this wave's 160-d slice (16q x 32k) ---
    v8f s0 = {}, s1 = {};
    const unsigned short* kl = &Kt[buf][0][d0];
    #pragma unroll
    for (int c = 0; c < 5; ++c) {
      v16bf b0 = load_bfrag(kl + c * 32, 640, lane);
      v16bf b1 = load_bfrag(kl + 16 * 640 + c * 32, 640, lane);
      s0 = wmma_bf16(aq[c], b0, s0);
      s1 = wmma_bf16(aq[c], b1, s1);
    }
    #pragma unroll
    for (int v = 0; v < 8; ++v) {
      sred[wave][0][v][lane] = s0[v];
      sred[wave][1][v][lane] = s1[v];
    }
    __syncthreads();

    // --- cross-wave reduce + online softmax (one thread per q-row) ---
    if (tid < 16) {
      const int r = tid, vv = r & 7, lbase = (r >= 8) ? 16 : 0;
      float sv[32]; float mx = -1e30f;
      #pragma unroll
      for (int j = 0; j < 32; ++j) {
        const int tile = j >> 4, nn = (j & 15) + lbase;
        float x = sred[0][tile][vv][nn] + sred[1][tile][vv][nn] +
                  sred[2][tile][vv][nn] + sred[3][tile][vv][nn];
        x *= 0.0395284708f;                 // 1/sqrt(640)
        sv[j] = x; mx = fmaxf(mx, x);
      }
      const float mo = m_run[r];
      const float mn = fmaxf(mo, mx);
      const float al = __expf(mo - mn);
      float sum = 0.0f;
      #pragma unroll
      for (int j = 0; j < 32; ++j) {
        const float p = __expf(sv[j] - mn);
        sum += p;
        Pb[r][j] = f2bf(p);
      }
      l_run[r] = l_run[r] * al + sum;
      m_run[r] = mn;
      alpha_s[r] = al;
    }
    __syncthreads();

    // --- rescale O and accumulate P @ V (V from LDS) ---
    #pragma unroll
    for (int v = 0; v < 8; ++v) {
      const float al = alpha_s[v + ((lane >= 16) ? 8 : 0)];
      #pragma unroll
      for (int t2 = 0; t2 < 10; ++t2) o[t2][v] *= al;
    }
    v16bf pf = load_afrag(&Pb[0][0], 32, lane);
    #pragma unroll
    for (int t2 = 0; t2 < 10; ++t2) {
      v16bf bv = load_bfrag(&Vt[buf][d0 + t2 * 16][0], 32, lane);
      o[t2] = wmma_bf16(pf, bv, o[t2]);
    }
  }

  #pragma unroll
  for (int v = 0; v < 8; ++v) {
    const int m = v + ((lane >= 16) ? 8 : 0);
    const float inv = 1.0f / l_run[m];
    const size_t rowoff = (size_t)(b * SEQ + q0 + m) * D_MODEL + head * D_HEAD + d0;
    #pragma unroll
    for (int t2 = 0; t2 < 10; ++t2)
      attn_bf[rowoff + t2 * 16 + (lane & 15)] = f2bf(o[t2][v] * inv);
  }
}

// ---------------------------------------------------------------------------
// Kernel D: out = attn_out @ out_w + out_b, B from pre-transposed bf16
// weights. No LDS, no barriers. grid = (256, 80), block = 128.
// ---------------------------------------------------------------------------
__global__ void __launch_bounds__(128)
k_out_proj(const unsigned short* __restrict__ attn_bf,
           const unsigned short* __restrict__ outw_bfT,
           const float* __restrict__ outb,
           float* __restrict__ out)
{
  const int m0   = blockIdx.x * 16;
  const int wave = threadIdx.x >> 5;
  const int lane = threadIdx.x & 31;
  const int n0   = blockIdx.y * 64 + wave * 16;

  const unsigned short* arow = attn_bf + (size_t)m0 * D_MODEL;
  const unsigned short* brow = outw_bfT + (size_t)n0 * D_MODEL;

  v8f acc = {};
  for (int k0 = 0; k0 < D_MODEL; k0 += 64) {
    __builtin_prefetch(brow + k0 + 256, 0, 0);
    v16bf a0 = load_afrag(arow + k0, D_MODEL, lane);
    v16bf b0 = load_bfrag(brow + k0, D_MODEL, lane);
    acc = wmma_bf16(a0, b0, acc);
    v16bf a1 = load_afrag(arow + k0 + 32, D_MODEL, lane);
    v16bf b1 = load_bfrag(brow + k0 + 32, D_MODEL, lane);
    acc = wmma_bf16(a1, b1, acc);
  }
  const int n  = lane & 15;
  const int mh = (lane >> 4) << 3;
  const float bn = outb[n0 + n];
  #pragma unroll
  for (int v = 0; v < 8; ++v)
    out[(size_t)(m0 + v + mh) * D_MODEL + n0 + n] = acc[v] + bn;
}

// ---------------------------------------------------------------------------
extern "C" void kernel_launch(void* const* d_in, const int* in_sizes, int n_in,
                              void* d_out, int out_size, void* d_ws, size_t ws_size,
                              hipStream_t stream) {
  (void)in_sizes; (void)n_in; (void)out_size; (void)ws_size;
  const float* h    = (const float*)d_in[0];
  const float* Wdkv = (const float*)d_in[1];
  const float* bdkv = (const float*)d_in[2];
  const float* Wdq  = (const float*)d_in[3];
  const float* bdq  = (const float*)d_in[4];
  const float* Wuk  = (const float*)d_in[5];
  const float* buk  = (const float*)d_in[6];
  const float* Wuv  = (const float*)d_in[7];
  const float* buv  = (const float*)d_in[8];
  const float* Wuq  = (const float*)d_in[9];
  const float* buq  = (const float*)d_in[10];
  const float* Wqr  = (const float*)d_in[11];
  const float* bqr  = (const float*)d_in[12];
  const float* Wkr  = (const float*)d_in[13];
  const float* bkr  = (const float*)d_in[14];
  const float* outw = (const float*)d_in[15];
  const float* outb = (const float*)d_in[16];

  float* out      = (float*)d_out;
  float* ckv_out  = out + (size_t)NROWS * D_MODEL;
  float* krot_out = ckv_out + (size_t)NROWS * D_KV;

  char* p = (char*)d_ws;
  unsigned short* ckv_bf   = (unsigned short*)p; p += (size_t)NROWS * D_KV * 2;
  unsigned short* cq_bf    = (unsigned short*)p; p += (size_t)NROWS * D_KV * 2;
  unsigned short* qbf      = (unsigned short*)p; p += (size_t)NROWS * D_MODEL * 2;
  unsigned short* kbf      = (unsigned short*)p; p += (size_t)NROWS * D_MODEL * 2;
  unsigned short* vT       = (unsigned short*)p; p += (size_t)NROWS * D_MODEL * 2;
  unsigned short* attn_bf  = (unsigned short*)p; p += (size_t)NROWS * D_MODEL * 2;
  unsigned short* outw_bfT = (unsigned short*)p; p += (size_t)D_MODEL * D_MODEL * 2;

  k_wprep<<<dim3(D_MODEL / 32, D_MODEL / 32), 256, 0, stream>>>(outw, outw_bfT);
  k_down_proj<<<dim3(NROWS / 16, 4), 128, 0, stream>>>(
      h, Wdkv, bdkv, Wdq, bdq, ckv_out, ckv_bf, cq_bf);
  k_up_proj_rope<<<dim3(NROWS / 16, 240), 128, 0, stream>>>(
      ckv_bf, cq_bf, Wuk, buk, Wuv, buv, Wuq, buq, Wqr, bqr, Wkr, bkr,
      qbf, kbf, vT, krot_out);
  k_attention<<<dim3(SEQ / 16, N_HEADS, BATCH), 128, 0, stream>>>(
      qbf, kbf, vT, attn_bf);
  k_out_proj<<<dim3(NROWS / 16, D_MODEL / 64), 128, 0, stream>>>(
      attn_bf, outw_bfT, outb, out);
}
